// SAR73_66228395705102
// MI455X (gfx1250) — matrix operand order repaired and checked by reading, not compiled, not run on hardware
//
#include <hip/hip_runtime.h>

// ===========================================================================
// Transformer autoencoder forward for MI455X (gfx1250, wave32, WMMA).
// All matmuls use v_wmma_f32_16x16x32_bf16 (bf16 in, f32 accumulate).
// Workspace layout (requires ws_size >= 320 MB):
//   [0,   64MB)  z          : fp32 [32768 x 512]
//   [64, 256MB)  A region   : q|k|v fp32 (3 x 64MB)  OR  ff-hidden bf16 (128MB)
//                             OR xh fp32 (8MB)
//   [256,320MB)  t region   : gemm epilogue temp / decoder LN / xp (8MB)
// ===========================================================================

typedef __attribute__((ext_vector_type(16))) __bf16 v16bf;
typedef __attribute__((ext_vector_type(8)))  float  v8f;

#define N_TOK   32768
#define DMODEL  512
#define DFF     2048
#define NLAYER  4
#define NHEAD   8
#define NPATCH  16

// ---- bf16 helpers: native converts (v_cvt_*_bf16), raw ushort storage -----
__device__ __forceinline__ unsigned short f2bf(float f) {
  union { __bf16 h; unsigned short u; } cv;
  cv.h = (__bf16)f;
  return cv.u;
}
__device__ __forceinline__ unsigned pack2bf(float a, float b) {
  return (unsigned)f2bf(a) | ((unsigned)f2bf(b) << 16);
}

union FragU { unsigned u[8]; v16bf v; };

// A fragment (16x32 bf16, contiguous k order, not the ISA table order): lane = hi*16 + (m%16)
//   element e -> K = hi*16 + e; stored as 8 packed uints.
__device__ __forceinline__ v16bf load_a_frag(const unsigned short* base, int lda,
                                             int m, int k0, int lane) {
  FragU r;
  const unsigned* p = (const unsigned*)(base + m * lda + k0);
  const int h4 = (lane >> 4) << 3;  // hi*8 pairs
  #pragma unroll
  for (int j = 0; j < 4; ++j) r.u[j]     = p[h4 + j];
  #pragma unroll
  for (int j = 0; j < 4; ++j) r.u[4 + j] = p[4 + h4 + j];
  return r.v;
}
// B fragment (32x16): element e -> K = hi*16 + e, N = lane&15.
// Bs is staged TRANSPOSED: Bs[n][k] = B[k][n].
__device__ __forceinline__ v16bf load_b_frag(const unsigned short* base, int ldb,
                                             int n, int k0, int lane) {
  FragU r;
  const unsigned* p = (const unsigned*)(base + n * ldb + k0 + ((lane >> 4) << 4));
  #pragma unroll
  for (int j = 0; j < 8; ++j) r.u[j] = p[j];
  return r.v;
}
#define WMMA_BF16(a, b, c) \
  __builtin_amdgcn_wmma_f32_16x16x32_bf16(false, (a), false, (b), (short)0, (c), false, false)

// ===========================================================================
// Generic GEMM: C[M,N] = epi(A[M,K] @ W[K,N] + bias[N])
//   A: fp32 (Af) or bf16 (Ah).  C: fp32 (Cf) or bf16 (Ch).
//   row_add: optional [64 x N] added by (row % 64)  (spatial encoding).
//   act_gelu: exact GELU.
// Tile: BM=128, BN=64, BK=64. 256 threads = 8 waves (4x2), wave tile 32x32.
// 8 WMMAs per wave per barrier pair.
// ===========================================================================
#define BM 128
#define BN 64
#define BK 64
#define LDS_K 66   // BK + 2 pad -> 33-uint row stride, conflict-free

__global__ __launch_bounds__(256) void gemm_kernel(
    const float* __restrict__ Af, const unsigned short* __restrict__ Ah,
    const float* __restrict__ W, const float* __restrict__ bias,
    const float* __restrict__ row_add,
    float* __restrict__ Cf, unsigned short* __restrict__ Ch,
    int act_gelu, int M, int N, int K)
{
  __shared__ alignas(16) unsigned short As[BM * LDS_K];
  __shared__ alignas(16) unsigned short Bs[BN * LDS_K];

  const int t = threadIdx.x;
  const int lane = t & 31, wave = t >> 5;
  const int wm = wave >> 1, wn = wave & 1;
  const int m0 = blockIdx.y * BM, n0 = blockIdx.x * BN;

  v8f acc[2][2] = {};

  for (int k0 = 0; k0 < K; k0 += BK) {
    // ---- prefetch next k-block (global_prefetch_b8) ----
    if (k0 + BK < K) {
      const int r = t >> 1;
      if ((t & 1) == 0) {
        const void* pa = Af ? (const void*)(Af + (size_t)(m0 + r) * K + k0 + BK)
                            : (const void*)(Ah + (size_t)(m0 + r) * K + k0 + BK);
        __builtin_prefetch(pa, 0, 0);
      } else if (r < BK) {
        __builtin_prefetch(W + (size_t)(k0 + BK + r) * N + n0, 0, 0);
      }
    }
    // ---- stage A tile (convert fp32 -> bf16, or copy bf16 pairs) ----
    if (Af) {
      #pragma unroll
      for (int it = 0; it < 8; ++it) {
        int vid = t + it * 256;              // 2048 float4 = 128x64
        int r = vid >> 4, c4 = (vid & 15) << 2;
        float4 f = *(const float4*)(Af + (size_t)(m0 + r) * K + k0 + c4);
        unsigned* dst = (unsigned*)(As + r * LDS_K + c4);
        dst[0] = pack2bf(f.x, f.y);
        dst[1] = pack2bf(f.z, f.w);
      }
    } else {
      #pragma unroll
      for (int it = 0; it < 16; ++it) {
        int pid = t + it * 256;              // 4096 bf16-pairs
        int r = pid >> 5, c2 = (pid & 31) << 1;
        unsigned v = *(const unsigned*)(Ah + (size_t)(m0 + r) * K + k0 + c2);
        *(unsigned*)(As + r * LDS_K + c2) = v;
      }
    }
    // ---- stage W tile transposed: Bs[n][k] ----
    #pragma unroll
    for (int it = 0; it < 4; ++it) {
      int vid = t + it * 256;                // 1024 float4 = 64x64
      int kr = vid >> 4, c4 = (vid & 15) << 2;
      float4 f = *(const float4*)(W + (size_t)(k0 + kr) * N + n0 + c4);
      Bs[(c4 + 0) * LDS_K + kr] = f2bf(f.x);
      Bs[(c4 + 1) * LDS_K + kr] = f2bf(f.y);
      Bs[(c4 + 2) * LDS_K + kr] = f2bf(f.z);
      Bs[(c4 + 3) * LDS_K + kr] = f2bf(f.w);
    }
    __syncthreads();

    #pragma unroll
    for (int kk = 0; kk < 2; ++kk) {
      v16bf af[2], bf[2];
      #pragma unroll
      for (int mt = 0; mt < 2; ++mt)
        af[mt] = load_a_frag(As, LDS_K, wm * 32 + mt * 16 + (lane & 15), kk * 32, lane);
      #pragma unroll
      for (int nt = 0; nt < 2; ++nt)
        bf[nt] = load_b_frag(Bs, LDS_K, wn * 32 + nt * 16 + (lane & 15), kk * 32, lane);
      #pragma unroll
      for (int mt = 0; mt < 2; ++mt)
        #pragma unroll
        for (int nt = 0; nt < 2; ++nt)
          acc[mt][nt] = WMMA_BF16(af[mt], bf[nt], acc[mt][nt]);
    }
    __syncthreads();
  }

  // ---- epilogue (C layout: VGPR r -> M = r | 8+r by lane half) ----
  const int hi = lane >> 4, nl = lane & 15;
  #pragma unroll
  for (int mt = 0; mt < 2; ++mt)
    #pragma unroll
    for (int nt = 0; nt < 2; ++nt) {
      const int n = n0 + wn * 32 + nt * 16 + nl;
      #pragma unroll
      for (int r = 0; r < 8; ++r) {
        const int m = m0 + wm * 32 + mt * 16 + (hi ? 8 + r : r);
        float v = acc[mt][nt][r] + bias[n];
        if (row_add) v += row_add[(size_t)(m & 63) * N + n];
        if (act_gelu) v = 0.5f * v * (1.0f + erff(v * 0.70710678118f));
        if (Cf) Cf[(size_t)m * N + n] = v;
        else    Ch[(size_t)m * N + n] = f2bf(v);
      }
    }
}

// ===========================================================================
// Attention: one WG (128 thr = 4 waves) per (bp, head).
// scores = (Qh Kh^T)/8 -> softmax -> zero diag -> renorm -> s_all, attn = S Vh
// attn_out aliases q (disjoint per-WG slices; Q fully read before writes).
// ===========================================================================
__global__ __launch_bounds__(128) void attn_kernel(
    const float* __restrict__ q, const float* __restrict__ k,
    const float* __restrict__ v, float* __restrict__ attn_out,
    float* __restrict__ s_all, int layer)
{
  __shared__ alignas(16) unsigned short Qs[64 * 66];
  __shared__ alignas(16) unsigned short Ks[64 * 66];
  __shared__ alignas(16) unsigned short Vt[64 * 66];   // Vt[n][kk] = V[kk][n]
  __shared__ alignas(16) unsigned short Ss[64 * 66];
  __shared__ float scr[64 * 65];

  const int t = threadIdx.x, lane = t & 31, wave = t >> 5;
  const int bp = blockIdx.y, h = blockIdx.x;
  const size_t base = (size_t)bp * 64 * DMODEL + (size_t)h * 64;

  #pragma unroll 4
  for (int it = 0; it < 32; ++it) {
    int lin = t + it * 128; int r = lin >> 6, c = lin & 63;
    Qs[r * 66 + c] = f2bf(q[base + (size_t)r * DMODEL + c]);
    Ks[r * 66 + c] = f2bf(k[base + (size_t)r * DMODEL + c]);
    Vt[c * 66 + r] = f2bf(v[base + (size_t)r * DMODEL + c]);
  }
  __syncthreads();

  const int ml = wave * 16 + (lane & 15);
  const int hi = lane >> 4, nl = lane & 15;

  { // scores strip: rows [16*wave, 16*wave+16), all 64 cols
    v16bf a0 = load_a_frag(Qs, 66, ml, 0, lane);
    v16bf a1 = load_a_frag(Qs, 66, ml, 32, lane);
    #pragma unroll
    for (int nt = 0; nt < 4; ++nt) {
      v16bf b0 = load_b_frag(Ks, 66, nt * 16 + nl, 0, lane);
      v16bf b1 = load_b_frag(Ks, 66, nt * 16 + nl, 32, lane);
      v8f acc = {};
      acc = WMMA_BF16(a0, b0, acc);
      acc = WMMA_BF16(a1, b1, acc);
      #pragma unroll
      for (int r = 0; r < 8; ++r) {
        int mm = wave * 16 + (hi ? 8 + r : r);
        scr[mm * 65 + nt * 16 + nl] = acc[r] * 0.125f;
      }
    }
  }
  __syncthreads();

  if (t < 64) { // softmax + diag mask + renorm, one row per thread
    const int r = t;
    float mx = -3.4e38f;
    for (int c = 0; c < 64; ++c) mx = fmaxf(mx, scr[r * 65 + c]);
    float sum = 0.f;
    for (int c = 0; c < 64; ++c) {
      float e = expf(scr[r * 65 + c] - mx);
      scr[r * 65 + c] = e; sum += e;
    }
    const float inv  = 1.0f / sum;
    const float sum2 = (sum - scr[r * 65 + r]) * inv;   // masked prob mass
    const float rn   = inv / (sum2 + 1e-6f);
    for (int c = 0; c < 64; ++c) {
      float p = (c == r) ? 0.f : scr[r * 65 + c] * rn;
      scr[r * 65 + c] = p;
      Ss[r * 66 + c] = f2bf(p);
    }
  }
  __syncthreads();

  { // coalesced s_all write: [B, L, P, H, 64, 64]
    float* sp = s_all +
      (((((size_t)(bp >> 4) * NLAYER + layer) * NPATCH + (bp & 15)) * NHEAD + h) << 12);
    #pragma unroll 4
    for (int it = 0; it < 32; ++it) {
      int lin = t + it * 128; int r = lin >> 6, c = lin & 63;
      sp[lin] = scr[r * 65 + c];
    }
  }

  { // attn = S @ V
    v16bf a0 = load_a_frag(Ss, 66, ml, 0, lane);
    v16bf a1 = load_a_frag(Ss, 66, ml, 32, lane);
    #pragma unroll
    for (int nt = 0; nt < 4; ++nt) {
      v16bf b0 = load_b_frag(Vt, 66, nt * 16 + nl, 0, lane);
      v16bf b1 = load_b_frag(Vt, 66, nt * 16 + nl, 32, lane);
      v8f acc = {};
      acc = WMMA_BF16(a0, b0, acc);
      acc = WMMA_BF16(a1, b1, acc);
      #pragma unroll
      for (int r = 0; r < 8; ++r) {
        int mm = wave * 16 + (hi ? 8 + r : r);
        attn_out[base + (size_t)mm * DMODEL + nt * 16 + nl] = acc[r];
      }
    }
  }
}

// ===========================================================================
// out[row] = LN(x[row] + (y ? y[row] : 0)) over D=512. One row per block.
// ===========================================================================
__global__ __launch_bounds__(256) void add_ln_kernel(
    const float* __restrict__ x, const float* __restrict__ y,
    const float* __restrict__ g, const float* __restrict__ b,
    float* __restrict__ out)
{
  __shared__ float rs[256], rq[256];
  const int t = threadIdx.x;
  const size_t row = (size_t)blockIdx.x * DMODEL;
  float v0 = x[row + t], v1 = x[row + t + 256];
  if (y) { v0 += y[row + t]; v1 += y[row + t + 256]; }
  rs[t] = v0 + v1;
  rq[t] = v0 * v0 + v1 * v1;
  __syncthreads();
  #pragma unroll
  for (int off = 128; off > 0; off >>= 1) {
    if (t < off) { rs[t] += rs[t + off]; rq[t] += rq[t + off]; }
    __syncthreads();
  }
  const float mean = rs[0] * (1.0f / DMODEL);
  const float var  = rq[0] * (1.0f / DMODEL) - mean * mean;
  const float rinv = rsqrtf(var + 1e-5f);
  out[row + t]       = g[t]       * (v0 - mean) * rinv + b[t];
  out[row + t + 256] = g[t + 256] * (v1 - mean) * rinv + b[t + 256];
}

// ===========================================================================
// Patch transposes: x[B,W,I] <-> xp[B*P*I, S]  (64x64 LDS tile, pad 65)
// ===========================================================================
__global__ __launch_bounds__(256) void patch_in_kernel(
    const float* __restrict__ x, float* __restrict__ xp)
{
  __shared__ float tile[64 * 65];
  const int bp = blockIdx.x, t = threadIdx.x;
  const float* src = x + (size_t)(bp >> 4) * 65536 + (size_t)(bp & 15) * 4096;
  #pragma unroll 4
  for (int it = 0; it < 16; ++it) {
    int lin = t + it * 256; int s = lin >> 6, i = lin & 63;
    tile[s * 65 + i] = src[lin];                  // read coalesced over i
  }
  __syncthreads();
  float* dst = xp + (size_t)bp * 4096;            // dst[i*64 + s]
  #pragma unroll 4
  for (int it = 0; it < 16; ++it) {
    int lin = t + it * 256; int i = lin >> 6, s = lin & 63;
    dst[lin] = tile[s * 65 + i];                  // write coalesced over s
  }
}

__global__ __launch_bounds__(256) void patch_out_kernel(
    const float* __restrict__ xh, float* __restrict__ xhat)
{
  __shared__ float tile[64 * 65];
  const int bp = blockIdx.x, t = threadIdx.x;
  const float* src = xh + (size_t)bp * 4096;      // src[i*64 + s]
  #pragma unroll 4
  for (int it = 0; it < 16; ++it) {
    int lin = t + it * 256; int i = lin >> 6, s = lin & 63;
    tile[s * 65 + i] = src[lin];
  }
  __syncthreads();
  float* dst = xhat + (size_t)(bp >> 4) * 65536 + (size_t)(bp & 15) * 4096;
  #pragma unroll 4
  for (int it = 0; it < 16; ++it) {
    int lin = t + it * 256; int s = lin >> 6, i = lin & 63;
    dst[lin] = tile[s * 65 + i];                  // write coalesced over i
  }
}

// ===========================================================================
// Host orchestration
// ===========================================================================
static inline void launch_gemm(const float* Af, const unsigned short* Ah,
                               const float* W, const float* bias, const float* row_add,
                               float* Cf, unsigned short* Ch, int act,
                               int M, int N, int K, hipStream_t s) {
  dim3 grid(N / BN, M / BM);
  gemm_kernel<<<grid, 256, 0, s>>>(Af, Ah, W, bias, row_add, Cf, Ch, act, M, N, K);
}

extern "C" void kernel_launch(void* const* d_in, const int* in_sizes, int n_in,
                              void* d_out, int out_size, void* d_ws, size_t ws_size,
                              hipStream_t stream) {
  (void)in_sizes; (void)n_in; (void)out_size; (void)ws_size;

  const float* x      = (const float*)d_in[0];
  const float* emb_w  = (const float*)d_in[1];
  const float* emb_b  = (const float*)d_in[2];
  const float* se     = (const float*)d_in[3];
  const float* dec_g  = (const float*)d_in[4];
  const float* dec_b  = (const float*)d_in[5];
  const float* dec_w  = (const float*)d_in[6];
  const float* dec_wb = (const float*)d_in[7];
  #define LP(l, j) ((const float*)d_in[8 + (l) * 16 + (j)])

  char* w = (char*)d_ws;
  float*          z  = (float*)w;                                   // 64MB
  float*          qb = (float*)(w + ((size_t)64  << 20));           // A region
  float*          kb = qb + (size_t)N_TOK * DMODEL;
  float*          vb = kb + (size_t)N_TOK * DMODEL;
  unsigned short* hb = (unsigned short*)qb;                         // ff hidden
  float*          xh = qb;                                          // decoder out
  float*          tb = (float*)(w + ((size_t)256 << 20));           // t region
  float*          xp = tb;

  float* xhat = (float*)d_out;
  float* sall = (float*)d_out + (size_t)N_TOK * 64;                 // after x_hat

  // --- embed: xp = patch-transpose(x); z = xp @ emb_w + emb_b + se ---
  patch_in_kernel<<<512, 256, 0, stream>>>(x, xp);
  launch_gemm(xp, nullptr, emb_w, emb_b, se, z, nullptr, 0,
              N_TOK, DMODEL, 64, stream);

  for (int l = 0; l < NLAYER; ++l) {
    launch_gemm(z, nullptr, LP(l, 0), LP(l, 1), nullptr, qb, nullptr, 0,
                N_TOK, DMODEL, DMODEL, stream);                     // Q
    launch_gemm(z, nullptr, LP(l, 2), LP(l, 3), nullptr, kb, nullptr, 0,
                N_TOK, DMODEL, DMODEL, stream);                     // K
    launch_gemm(z, nullptr, LP(l, 4), LP(l, 5), nullptr, vb, nullptr, 0,
                N_TOK, DMODEL, DMODEL, stream);                     // V
    attn_kernel<<<dim3(NHEAD, 512), 128, 0, stream>>>(qb, kb, vb, qb, sall, l);
    launch_gemm(qb, nullptr, LP(l, 6), LP(l, 7), nullptr, tb, nullptr, 0,
                N_TOK, DMODEL, DMODEL, stream);                     // O proj
    add_ln_kernel<<<N_TOK, 256, 0, stream>>>(z, tb, LP(l, 8), LP(l, 9), z);
    launch_gemm(z, nullptr, LP(l, 12), LP(l, 13), nullptr, nullptr, hb, 1,
                N_TOK, DFF, DMODEL, stream);                        // FF1 + GELU (bf16 out)
    launch_gemm(nullptr, hb, LP(l, 14), LP(l, 15), nullptr, tb, nullptr, 0,
                N_TOK, DMODEL, DFF, stream);                        // FF2
    add_ln_kernel<<<N_TOK, 256, 0, stream>>>(z, tb, LP(l, 10), LP(l, 11), z);
  }

  // --- decoder: xhat = unpatch(LN(z) @ dec_w + dec_wb) ---
  add_ln_kernel<<<N_TOK, 256, 0, stream>>>(z, nullptr, dec_g, dec_b, tb);
  launch_gemm(tb, nullptr, dec_w, dec_wb, nullptr, xh, nullptr, 0,
              N_TOK, 64, DMODEL, stream);
  patch_out_kernel<<<512, 256, 0, stream>>>(xh, xhat);
  #undef LP
}
